// MultiHeadMetaToDataPredictor_65755949302207
// MI455X (gfx1250) — matrix operand order repaired and checked by reading, not compiled, not run on hardware
//
#include <hip/hip_runtime.h>

// CDNA5 / gfx1250 routed-MoE MLP. bf16 WMMA (v_wmma_f32_16x16x32_bf16),
// wave32, 8 waves/block, full 4-layer pipeline resident in LDS.

typedef __attribute__((ext_vector_type(16))) __bf16 v16bf;
typedef __attribute__((ext_vector_type(8)))  __bf16 v8bf;
typedef __attribute__((ext_vector_type(8)))  float  v8f;

namespace {
constexpr int NH    = 24;
constexpr int DIN   = 136;
constexpr int K1P   = 160;     // DIN zero-padded to multiple of 32
constexpr int D1    = 512;
constexpr int D2    = 256;
constexpr int D3    = 128;
constexpr int DOUTN = 1024;
constexpr int BATCH = 8192;
constexpr int MAXTILES = BATCH/16 + NH;   // 536

// int32 meta region at start of d_ws
constexpr int M_COUNTS = 0;      // [24]
constexpr int M_OFFS   = 24;     // [25]
constexpr int M_CURSOR = 49;     // [24]
constexpr int M_NTILES = 73;     // [1]
constexpr int M_THEAD  = 74;     // [536]
constexpr int M_TFIRST = 610;    // [536]
constexpr int M_ORDER  = 1146;   // [8192]
constexpr int META_BYTES = 37376; // 9344 ints, 32B aligned

// bf16 weight image offsets (elements), fragment layout [nt][kt][n16][kk32]
constexpr long long S1 = (long long)K1P*D1;     // 81920
constexpr long long S2 = (long long)D1*D2;      // 131072
constexpr long long S3 = (long long)D2*D3;      // 32768
constexpr long long S4 = (long long)D3*DOUTN;   // 131072
constexpr long long OFF1 = 0;
constexpr long long OFF2 = OFF1 + (long long)NH*S1;
constexpr long long OFF3 = OFF2 + (long long)NH*S2;
constexpr long long OFF4 = OFF3 + (long long)NH*S3;
} // namespace

// ---------------- routing ----------------
__global__ void k_route_init(int* meta) {
  int t = threadIdx.x;
  if (t < NH) meta[M_COUNTS + t] = 0;
  if (t == 0) meta[M_NTILES] = 0;
}

__global__ void k_count(const int* __restrict__ head_idx, int* meta) {
  int i = blockIdx.x * blockDim.x + threadIdx.x;
  if (i < BATCH) atomicAdd(&meta[M_COUNTS + head_idx[i]], 1);
}

__global__ void k_scan(int* meta) {
  if (threadIdx.x != 0 || blockIdx.x != 0) return;
  int acc = 0, nt = 0;
  for (int h = 0; h < NH; ++h) {
    meta[M_OFFS + h]   = acc;
    meta[M_CURSOR + h] = acc;
    int c = meta[M_COUNTS + h];
    for (int t = 0; t < c; t += 16) {
      meta[M_THEAD  + nt] = h;
      meta[M_TFIRST + nt] = acc + t;
      ++nt;
    }
    acc += c;
  }
  meta[M_OFFS + NH] = acc;
  meta[M_NTILES] = nt;
}

__global__ void k_scatter(const int* __restrict__ head_idx, int* meta) {
  int i = blockIdx.x * blockDim.x + threadIdx.x;
  if (i < BATCH) {
    int h = head_idx[i];
    int pos = atomicAdd(&meta[M_CURSOR + h], 1);
    meta[M_ORDER + pos] = i;
  }
}

// ------------- weight fp32 -> bf16 fragment image -------------
// dst layout per head: [nt][kt][n:16][kk:32], K zero-padded to Kpad.
__global__ void k_convert_w(const float* __restrict__ W, __bf16* __restrict__ dst,
                            int Ksrc, int Kpad, int N) {
  int per = Kpad * N;
  int ktCount = Kpad >> 5;
  long long total = (long long)NH * per;
  for (long long d = (long long)blockIdx.x * blockDim.x + threadIdx.x;
       d < total; d += (long long)gridDim.x * blockDim.x) {
    int head = (int)(d / per);
    int r    = (int)(d % per);
    int kk = r & 31;
    int n  = (r >> 5) & 15;
    int kt = (r >> 9) % ktCount;
    int nt = (r >> 9) / ktCount;
    int K    = kt * 32 + kk;
    int Ncol = nt * 16 + n;
    float v = (K < Ksrc) ? W[(long long)head * Ksrc * N + (long long)K * N + Ncol] : 0.f;
    dst[d] = (__bf16)v;
  }
}

// ---------------- WMMA helpers ----------------
// A fragment (16x32 bf16): lane -> row (lane&15); elements 0..7 = K offset 16*half,
// elements 8..15 = K offset 16*half+8 (two contiguous 16B chunks).
__device__ __forceinline__ v16bf load_a_frag(const __bf16* arow, int half) {
  v8bf lo = *(const v8bf*)(arow + 16 * half);
  v8bf hi = *(const v8bf*)(arow + 16 * half + 8);
  return __builtin_shufflevector(lo, hi, 0,1,2,3,4,5,6,7,8,9,10,11,12,13,14,15);
}

// relu(A(16xKpad) * Wt + bias) -> Aout (16xN bf16 in LDS)
__device__ __forceinline__ void layer_fwd(const __bf16* __restrict__ Ain, int Kpad,
                                          const __bf16* __restrict__ Wt,
                                          const float* __restrict__ bias,
                                          __bf16* __restrict__ Aout, int N,
                                          int wave, int lane) {
  const int ktCount = Kpad >> 5;
  const int ntCount = N >> 4;
  const int half = lane >> 4;
  const int ln   = lane & 15;
  const __bf16* ar = Ain + ln * Kpad;
  for (int nt = wave; nt < ntCount; nt += 8) {
    v8f acc = {};
    const __bf16* wb = Wt + (long long)nt * ktCount * 512 + ln * 32 + 16 * half;
    for (int kt = 0; kt < ktCount; ++kt) {
      v16bf a = load_a_frag(ar + kt * 32, half);
      v16bf b = *(const v16bf*)(wb + kt * 512);   // contiguous 32B per lane
      acc = __builtin_amdgcn_wmma_f32_16x16x32_bf16(false, a, false, b,
                                                    (short)0, acc, false, false);
    }
    float bv = bias[nt * 16 + ln];
#pragma unroll
    for (int i = 0; i < 8; ++i) {
      float v = acc[i] + bv;
      v = v > 0.f ? v : 0.f;
      Aout[(8 * half + i) * N + nt * 16 + ln] = (__bf16)v;
    }
  }
}

// final layer: A(16x128) * W4t + b4 -> f32 rows scattered to d_out
__device__ __forceinline__ void layer_out(const __bf16* __restrict__ Ain, int Kpad,
                                          const __bf16* __restrict__ Wt,
                                          const float* __restrict__ bias,
                                          float* __restrict__ out,
                                          const int* __restrict__ sidx,
                                          int wave, int lane) {
  const int ktCount = Kpad >> 5;
  const int ntCount = DOUTN >> 4;
  const int half = lane >> 4;
  const int ln   = lane & 15;
  const __bf16* ar = Ain + ln * Kpad;
  for (int nt = wave; nt < ntCount; nt += 8) {
    v8f acc = {};
    const __bf16* wb = Wt + (long long)nt * ktCount * 512 + ln * 32 + 16 * half;
    for (int kt = 0; kt < ktCount; ++kt) {
      v16bf a = load_a_frag(ar + kt * 32, half);
      v16bf b = *(const v16bf*)(wb + kt * 512);
      acc = __builtin_amdgcn_wmma_f32_16x16x32_bf16(false, a, false, b,
                                                    (short)0, acc, false, false);
    }
    float bv = bias[nt * 16 + ln];
#pragma unroll
    for (int i = 0; i < 8; ++i) {
      int m = 8 * half + i;
      int s = sidx[m];
      if (s >= 0) out[(long long)s * DOUTN + nt * 16 + ln] = acc[i] + bv;
    }
  }
}

// ---------------- main routed-MLP kernel ----------------
__global__ __launch_bounds__(256, 2) void k_moe_main(
    const float* __restrict__ x, const int* __restrict__ meta,
    const __bf16* __restrict__ wsbf,
    const float* __restrict__ b1, const float* __restrict__ b2,
    const float* __restrict__ b3, const float* __restrict__ b4,
    float* __restrict__ out) {
  __shared__ __align__(32) __bf16 A0[16 * K1P];
  __shared__ __align__(32) __bf16 A1[16 * D1];
  __shared__ __align__(32) __bf16 A2[16 * D2];
  __shared__ __align__(32) __bf16 A3[16 * D3];
  __shared__ int sidx[16];

  const int nTiles = meta[M_NTILES];
  if ((int)blockIdx.x >= nTiles) return;

  const int h      = meta[M_THEAD  + blockIdx.x];
  const int first  = meta[M_TFIRST + blockIdx.x];
  const int gend   = meta[M_OFFS + h + 1];
  int nvalid = gend - first; if (nvalid > 16) nvalid = 16;

  const int tid = threadIdx.x;
  if (tid < 16) sidx[tid] = (tid < nvalid) ? meta[M_ORDER + first + tid] : -1;
  __syncthreads();

  // gather + fp32->bf16 the 16 input rows, zero-padded to K1P
  for (int idx = tid; idx < 16 * K1P; idx += 256) {
    int r = idx / K1P, c = idx - r * K1P;
    int s = sidx[r];
    float v = (s >= 0 && c < DIN) ? x[(long long)s * DIN + c] : 0.f;
    A0[idx] = (__bf16)v;
  }
  __syncthreads();

  const int wave = tid >> 5, lane = tid & 31;

  layer_fwd(A0, K1P, wsbf + OFF1 + (long long)h * S1, b1 + h * D1, A1, D1, wave, lane);
  __syncthreads();
  layer_fwd(A1, D1,  wsbf + OFF2 + (long long)h * S2, b2 + h * D2, A2, D2, wave, lane);
  __syncthreads();
  layer_fwd(A2, D2,  wsbf + OFF3 + (long long)h * S3, b3 + h * D3, A3, D3, wave, lane);
  __syncthreads();
  layer_out(A3, D3,  wsbf + OFF4 + (long long)h * S4, b4 + h * DOUTN, out, sidx, wave, lane);
}

// ---------------- host entry ----------------
extern "C" void kernel_launch(void* const* d_in, const int* in_sizes, int n_in,
                              void* d_out, int out_size, void* d_ws, size_t ws_size,
                              hipStream_t stream) {
  const float* x  = (const float*)d_in[0];
  const float* W1 = (const float*)d_in[1]; const float* b1 = (const float*)d_in[2];
  const float* W2 = (const float*)d_in[3]; const float* b2 = (const float*)d_in[4];
  const float* W3 = (const float*)d_in[5]; const float* b3 = (const float*)d_in[6];
  const float* W4 = (const float*)d_in[7]; const float* b4 = (const float*)d_in[8];
  const int* head_idx = (const int*)d_in[9];

  int*    meta = (int*)d_ws;
  __bf16* wsbf = (__bf16*)((char*)d_ws + META_BYTES);

  // 1) route samples to heads
  k_route_init<<<1, 64, 0, stream>>>(meta);
  k_count    <<<BATCH / 256, 256, 0, stream>>>(head_idx, meta);
  k_scan     <<<1, 1, 0, stream>>>(meta);
  k_scatter  <<<BATCH / 256, 256, 0, stream>>>(head_idx, meta);

  // 2) weights -> bf16 WMMA-fragment image in workspace
  k_convert_w<<<2048, 256, 0, stream>>>(W1, wsbf + OFF1, DIN, K1P, D1);
  k_convert_w<<<2048, 256, 0, stream>>>(W2, wsbf + OFF2, D1,  D1,  D2);
  k_convert_w<<<2048, 256, 0, stream>>>(W3, wsbf + OFF3, D2,  D2,  D3);
  k_convert_w<<<2048, 256, 0, stream>>>(W4, wsbf + OFF4, D3,  D3,  DOUTN);

  // 3) routed 4-layer MLP, one block per (head, 16-sample tile)
  k_moe_main<<<MAXTILES, 256, 0, stream>>>(x, meta, wsbf, b1, b2, b3, b4,
                                           (float*)d_out);
}
